// Autoencoder_36077725286661
// MI455X (gfx1250) — compile-verified, matrix operand order checked by reading, not run on hardware
//
#include <hip/hip_runtime.h>
#include <hip/hip_bf16.h>

// ---------------------------------------------------------------------------
// GraphMAE-style GAT autoencoder forward for MI455X (gfx1250, wave32).
// Dense linears run on v_wmma_f32_16x16x32_bf16 (fp32 accumulate); edge
// softmax + scatter run as memory-bound atomic kernels (fits in 192MB L2).
// ---------------------------------------------------------------------------

typedef __attribute__((ext_vector_type(16))) __bf16 v16bf;
typedef __attribute__((ext_vector_type(8)))  float  v8f;

#define GN 100000
#define GE 1600000
#define GIN 64
#define GHD 128

// ---------------- WMMA GEMM: C[M x Ncol] = A_bf16[M x K] @ B, B given as
// Bt_bf16[Ncol x K] (transposed so the 32x16 B fragment is a contiguous load).
// One wave computes a 16x64 tile: 4 accumulators reuse each A fragment.
__global__ void k_gemm_bf16(const __bf16* __restrict__ A,
                            const __bf16* __restrict__ Bt,
                            float* __restrict__ C,
                            int M, int K, int Ncol) {
  int wv   = (int)((blockIdx.x * blockDim.x + threadIdx.x) >> 5);
  int lane = (int)(threadIdx.x & 31);
  int cg = Ncol >> 6;                       // number of 64-column groups
  int totalWaves = (M >> 4) * cg;
  if (wv >= totalWaves) return;
  int tm = wv / cg;
  int tn = wv % cg;
  int half = lane >> 4;                     // K-half selector (A/B 16-bit layout)
  int l16  = lane & 15;

  const __bf16* arow = A + (size_t)(tm * 16 + l16) * K + half * 16;
  const __bf16* bbase = Bt + (size_t)(tn * 64 + l16) * K + half * 16;

  v8f acc0 = {}, acc1 = {}, acc2 = {}, acc3 = {};
  for (int kk = 0; kk < K; kk += 32) {
    v16bf a  = *(const v16bf*)(arow + kk);
    v16bf b0 = *(const v16bf*)(bbase + kk);
    v16bf b1 = *(const v16bf*)(bbase + kk + (size_t)16 * K);
    v16bf b2 = *(const v16bf*)(bbase + kk + (size_t)32 * K);
    v16bf b3 = *(const v16bf*)(bbase + kk + (size_t)48 * K);
    acc0 = __builtin_amdgcn_wmma_f32_16x16x32_bf16(false, a, false, b0, (short)0, acc0, false, false);
    acc1 = __builtin_amdgcn_wmma_f32_16x16x32_bf16(false, a, false, b1, (short)0, acc1, false, false);
    acc2 = __builtin_amdgcn_wmma_f32_16x16x32_bf16(false, a, false, b2, (short)0, acc2, false, false);
    acc3 = __builtin_amdgcn_wmma_f32_16x16x32_bf16(false, a, false, b3, (short)0, acc3, false, false);
  }
  int row0 = tm * 16 + half * 8;            // C/D layout: VGPR i -> M = half*8 + i
  int col  = tn * 64 + l16;
#pragma unroll
  for (int i = 0; i < 8; ++i) {
    size_t r = (size_t)(row0 + i) * Ncol;
    C[r + col +  0] = acc0[i];
    C[r + col + 16] = acc1[i];
    C[r + col + 32] = acc2[i];
    C[r + col + 48] = acc3[i];
  }
}

// ---------------- Weight prep: W[K x Ncol] fp32 -> Wt[Ncol x K] bf16
__global__ void k_transpose_bf16(const float* __restrict__ W, __bf16* __restrict__ Wt,
                                 int K, int Ncol) {
  int idx = blockIdx.x * blockDim.x + threadIdx.x;
  if (idx >= K * Ncol) return;
  int n = idx / K, k = idx % K;
  Wt[idx] = (__bf16)W[(size_t)k * Ncol + n];
}

// ---------------- Elementwise helpers
__global__ void k_f32_to_bf16(const float* __restrict__ a, __bf16* __restrict__ o,
                              int n, float scale) {
  int i = blockIdx.x * blockDim.x + threadIdx.x;
  if (i < n) o[i] = (__bf16)(a[i] * scale);
}

__global__ void k_avg2_bf16(const float* __restrict__ a, const float* __restrict__ b,
                            __bf16* __restrict__ o, int n) {
  int i = blockIdx.x * blockDim.x + threadIdx.x;
  if (i < n) o[i] = (__bf16)(0.5f * (a[i] + b[i]));
}

__global__ void k_set_rows_bf16(const int* __restrict__ rows, int nrows,
                                const float* __restrict__ token,
                                __bf16* __restrict__ o, int cols) {
  int i = blockIdx.x * blockDim.x + threadIdx.x;
  if (i >= nrows * cols) return;
  int r = rows[i / cols];
  int c = i % cols;
  o[(size_t)r * cols + c] = (__bf16)token[c];
}

__global__ void k_accum(float* __restrict__ out, const float* __restrict__ add,
                        float beta, int n) {
  int i = blockIdx.x * blockDim.x + threadIdx.x;
  if (i < n) out[i] = out[i] * beta + add[i];
}

// ---------------- Attention logits: el/er[n,h] = dot(feat[n,h,:], al/ar[h,:])
__global__ void k_attn_logits(const float* __restrict__ feat,
                              const float* __restrict__ al, const float* __restrict__ ar,
                              float* __restrict__ el, float* __restrict__ er, int Nn) {
  int idx = blockIdx.x * blockDim.x + threadIdx.x;
  if (idx >= Nn * 2) return;
  int n = idx >> 1, h = idx & 1;
  const float* f  = feat + (size_t)n * GHD + h * 64;
  const float* wl = al + h * 64;
  const float* wr = ar + h * 64;
  float sl = 0.f, sr = 0.f;
#pragma unroll 8
  for (int d = 0; d < 64; ++d) { float v = f[d]; sl += v * wl[d]; sr += v * wr[d]; }
  el[idx] = sl; er[idx] = sr;
}

// order-preserving float<->uint (for atomic segment max); 0 encodes "-inf-ish"
__device__ __forceinline__ unsigned f2o(float f) {
  unsigned u = __float_as_uint(f);
  return (u & 0x80000000u) ? ~u : (u | 0x80000000u);
}
__device__ __forceinline__ float o2f(unsigned u) {
  return __uint_as_float((u & 0x80000000u) ? (u & 0x7fffffffu) : ~u);
}
__device__ __forceinline__ float leaky(float x) { return x > 0.f ? x : 0.2f * x; }

__global__ void k_edge_max(const int* __restrict__ src, const int* __restrict__ dst,
                           const float* __restrict__ el, const float* __restrict__ er,
                           unsigned* __restrict__ mord, int En) {
  int idx = blockIdx.x * blockDim.x + threadIdx.x;
  if (idx >= En * 2) return;
  int e = idx >> 1, h = idx & 1;
  float x = leaky(el[src[e] * 2 + h] + er[dst[e] * 2 + h]);
  atomicMax(&mord[dst[e] * 2 + h], f2o(x));
}

__global__ void k_edge_exp(const int* __restrict__ src, const int* __restrict__ dst,
                           const float* __restrict__ el, const float* __restrict__ er,
                           const unsigned* __restrict__ mord,
                           float* __restrict__ denom, float* __restrict__ ex, int En) {
  int idx = blockIdx.x * blockDim.x + threadIdx.x;
  if (idx >= En * 2) return;
  int e = idx >> 1, h = idx & 1;
  int d0 = dst[e];
  float x = leaky(el[src[e] * 2 + h] + er[d0 * 2 + h]);
  float m = o2f(mord[d0 * 2 + h]);
  if (!(m >= -3.4e38f && m <= 3.4e38f)) m = 0.f;   // isfinite guard (ref semantics)
  float v = __expf(x - m);
  ex[idx] = v;
  atomicAdd(&denom[d0 * 2 + h], v);
}

// out[dst, d] += (ex/denom) * feat[src, d]; thread per (edge, dim) -> coalesced
__global__ void k_edge_scatter(const int* __restrict__ src, const int* __restrict__ dst,
                               const float* __restrict__ ex, const float* __restrict__ denom,
                               const float* __restrict__ feat, float* __restrict__ out,
                               int total) {
  int idx = blockIdx.x * blockDim.x + threadIdx.x;
  if (idx >= total) return;
  int e = idx >> 7;         // /128
  int d = idx & 127;
  int h = d >> 6;
  int dn = dst[e];
  float a = ex[e * 2 + h] / denom[dn * 2 + h];
  atomicAdd(&out[(size_t)dn * GHD + d], a * feat[(size_t)src[e] * GHD + d]);
}

// out = elu(acc + bias); optional fp32 and bf16 destinations
__global__ void k_bias_elu(const float* __restrict__ acc, const float* __restrict__ bias,
                           float* __restrict__ outF, __bf16* __restrict__ outB, int total) {
  int idx = blockIdx.x * blockDim.x + threadIdx.x;
  if (idx >= total) return;
  float v = acc[idx] + bias[idx & 127];
  v = v > 0.f ? v : (__expf(v) - 1.0f);
  if (outF) outF[idx] = v;
  if (outB) outB[idx] = (__bf16)v;
}

// ---------------------------------------------------------------------------
extern "C" void kernel_launch(void* const* d_in, const int* in_sizes, int n_in,
                              void* d_out, int out_size, void* d_ws, size_t ws_size,
                              hipStream_t stream) {
  const float* x        = (const float*)d_in[0];
  const int*   src      = (const int*)d_in[1];
  const int*   dst      = (const int*)d_in[2];
  const int*   encMask  = (const int*)d_in[3];
  const int*   remask   = (const int*)d_in[4];
  const float* enc_fc0  = (const float*)d_in[5];
  const float* enc_al0  = (const float*)d_in[6];
  const float* enc_ar0  = (const float*)d_in[7];
  const float* enc_b0   = (const float*)d_in[8];
  const float* enc_fc1  = (const float*)d_in[9];
  const float* enc_al1  = (const float*)d_in[10];
  const float* enc_ar1  = (const float*)d_in[11];
  const float* enc_b1   = (const float*)d_in[12];
  const float* dec_fc0  = (const float*)d_in[13];
  const float* dec_al0  = (const float*)d_in[14];
  const float* dec_ar0  = (const float*)d_in[15];
  const float* dec_b0   = (const float*)d_in[16];
  const float* dec_fc1  = (const float*)d_in[17];
  const float* dec_al1  = (const float*)d_in[18];
  const float* dec_ar1  = (const float*)d_in[19];
  const float* dec_b1   = (const float*)d_in[20];
  const float* encTok   = (const float*)d_in[21];
  const float* decTok   = (const float*)d_in[22];
  const float* W_e2d    = (const float*)d_in[23];
  const float* W_d2c    = (const float*)d_in[24];

  const int Nn = in_sizes[0] / GIN;           // 100000
  const int Ee = in_sizes[1];                 // 1600000
  const int nmask   = in_sizes[3];            // 30000
  const int nremask = in_sizes[4] / 3;        // 50000

  // ---- workspace layout (256B aligned offsets) ----
  char* base = (char*)d_ws;
  size_t off = 0;
  auto alloc = [&](size_t bytes) { size_t o = off; off += (bytes + 255) & ~(size_t)255; return o; };
  __bf16*   pingB  = (__bf16*)(base + alloc((size_t)Nn * GHD * 2));
  __bf16*   pongB  = (__bf16*)(base + alloc((size_t)Nn * GHD * 2));
  float*    featF  = (float*)(base + alloc((size_t)Nn * GHD * 4));
  float*    accF   = (float*)(base + alloc((size_t)Nn * GHD * 4));
  float*    h0F    = (float*)(base + alloc((size_t)Nn * GHD * 4));
  float*    decSum = (float*)(base + alloc((size_t)Nn * GHD * 4));
  float*    origF  = (float*)(base + alloc((size_t)Nn * GIN * 4));
  float*    elF    = (float*)(base + alloc((size_t)Nn * 2 * 4));
  float*    erF    = (float*)(base + alloc((size_t)Nn * 2 * 4));
  unsigned* mordU  = (unsigned*)(base + alloc((size_t)Nn * 2 * 4));
  float*    denF   = (float*)(base + alloc((size_t)Nn * 2 * 4));
  float*    exF    = (float*)(base + alloc((size_t)Ee * 2 * 4));
  __bf16*   wEnc0  = (__bf16*)(base + alloc((size_t)GHD * GIN * 2));
  __bf16*   wEnc1  = (__bf16*)(base + alloc((size_t)GHD * GHD * 2));
  __bf16*   wDec0  = (__bf16*)(base + alloc((size_t)GHD * GIN * 2));
  __bf16*   wDec1  = (__bf16*)(base + alloc((size_t)GHD * GHD * 2));
  __bf16*   wE2d   = (__bf16*)(base + alloc((size_t)GIN * GHD * 2));
  __bf16*   wD2c   = (__bf16*)(base + alloc((size_t)GIN * GHD * 2));

  const int BLK = 256;
  auto grid = [&](long long n) { return (unsigned)((n + BLK - 1) / BLK); };

  // ---- weight prep: transpose + bf16 ----
  k_transpose_bf16<<<grid(64 * 128), BLK, 0, stream>>>(enc_fc0, wEnc0, 64, 128);
  k_transpose_bf16<<<grid(128 * 128), BLK, 0, stream>>>(enc_fc1, wEnc1, 128, 128);
  k_transpose_bf16<<<grid(64 * 128), BLK, 0, stream>>>(dec_fc0, wDec0, 64, 128);
  k_transpose_bf16<<<grid(128 * 128), BLK, 0, stream>>>(dec_fc1, wDec1, 128, 128);
  k_transpose_bf16<<<grid(128 * 64), BLK, 0, stream>>>(W_e2d, wE2d, 128, 64);
  k_transpose_bf16<<<grid(128 * 64), BLK, 0, stream>>>(W_d2c, wD2c, 128, 64);

  auto gemm = [&](const __bf16* A, const __bf16* Bt, float* C, int K, int Ncol) {
    int waves = (Nn / 16) * (Ncol / 64);
    k_gemm_bf16<<<grid((long long)waves * 32), BLK, 0, stream>>>(A, Bt, C, Nn, K, Ncol);
  };

  // One GAT layer: inB[N x K] bf16 -> outF/outB[N x 128]
  auto gat = [&](const __bf16* inB, int K, const __bf16* Wt,
                 const float* al, const float* ar, const float* bias,
                 float* outF, __bf16* outB) {
    gemm(inB, Wt, featF, K, GHD);                                             // feat = in @ fc
    k_attn_logits<<<grid((long long)Nn * 2), BLK, 0, stream>>>(featF, al, ar, elF, erF, Nn);
    hipMemsetAsync(mordU, 0, (size_t)Nn * 2 * 4, stream);                     // ordered(-inf)=0
    hipMemsetAsync(denF, 0, (size_t)Nn * 2 * 4, stream);
    hipMemsetAsync(accF, 0, (size_t)Nn * GHD * 4, stream);
    k_edge_max<<<grid((long long)Ee * 2), BLK, 0, stream>>>(src, dst, elF, erF, mordU, Ee);
    k_edge_exp<<<grid((long long)Ee * 2), BLK, 0, stream>>>(src, dst, elF, erF, mordU, denF, exF, Ee);
    k_edge_scatter<<<grid((long long)Ee * GHD), BLK, 0, stream>>>(src, dst, exF, denF, featF, accF,
                                                                  Ee * GHD);
    k_bias_elu<<<grid((long long)Nn * GHD), BLK, 0, stream>>>(accF, bias, outF, outB, Nn * GHD);
  };

  // ---- encoder ----
  k_f32_to_bf16<<<grid((long long)Nn * GIN), BLK, 0, stream>>>(x, pingB, Nn * GIN, 1.f);
  k_set_rows_bf16<<<grid((long long)nmask * GIN), BLK, 0, stream>>>(encMask, nmask, encTok, pingB, GIN);
  gat(pingB, GIN, wEnc0, enc_al0, enc_ar0, enc_b0, h0F, pongB);               // h0
  gat(pongB, GHD, wEnc1, enc_al1, enc_ar1, enc_b1, accF, nullptr);            // h1 (in accF)
  k_avg2_bf16<<<grid((long long)Nn * GHD), BLK, 0, stream>>>(h0F, accF, pingB, Nn * GHD); // Es
  gemm(pingB, wE2d, origF, GHD, GIN);                                         // origin_rep

  // ---- decoder: 3 remask iterations ----
  for (int it = 0; it < 3; ++it) {
    k_f32_to_bf16<<<grid((long long)Nn * GIN), BLK, 0, stream>>>(origF, pongB, Nn * GIN, 1.f);
    k_set_rows_bf16<<<grid((long long)nremask * GIN), BLK, 0, stream>>>(
        remask + (size_t)it * nremask, nremask, decTok, pongB, GIN);
    gat(pongB, GIN, wDec0, dec_al0, dec_ar0, dec_b0, accF, pingB);            // r0 (bf16 in ping)
    gat(pingB, GHD, wDec1, dec_al1, dec_ar1, dec_b1, accF, nullptr);          // r1 (in accF)
    k_accum<<<grid((long long)Nn * GHD), BLK, 0, stream>>>(decSum, accF, it == 0 ? 0.f : 1.f,
                                                           Nn * GHD);
  }

  // ---- Ds = mean(decodes); out = Ds @ W_d2c ----
  k_f32_to_bf16<<<grid((long long)Nn * GHD), BLK, 0, stream>>>(decSum, pingB, Nn * GHD,
                                                               1.0f / 3.0f);
  gemm(pingB, wD2c, (float*)d_out, GHD, GIN);
}
